// MGADN_11751030522433
// MI455X (gfx1250) — compile-verified, matrix operand order checked by reading, not run on hardware
//
#include <hip/hip_runtime.h>
#include <hip/hip_bf16.h>
#include <math.h>

// ---------------- problem constants (match reference) ----------------
#define IN_F 128
#define HF_F 64
#define NH_H 8
#define HD_D 8
#define L_LAYERS 4

typedef __attribute__((ext_vector_type(16))) _Float16 v16h;
typedef __attribute__((ext_vector_type(8)))  _Float16 v8h;
typedef __attribute__((ext_vector_type(8)))  float    v8f;

// ---------------- monotonic float <-> u32 key for atomic max ----------------
__device__ __forceinline__ unsigned fenc(float f) {
    unsigned u = __float_as_uint(f);
    return (u & 0x80000000u) ? ~u : (u | 0x80000000u);
}
__device__ __forceinline__ float fdec(unsigned k) {
    return (k & 0x80000000u) ? __uint_as_float(k ^ 0x80000000u)
                             : __uint_as_float(~k);
}
#define KEY_NEG_INF 0x007FFFFFu   // fenc(-inf)

// =====================================================================
// WMMA GEMM, K = 64 output columns fixed (all dense layers here are *->64).
//   out[n x 64] = act( (A[n x M] @ W[M x 64]) * rowscale + bias )
// One wave computes a 16x64 row stripe (4 accumulators); W is staged into
// LDS transposed ([col][k]) in f16, so each B fragment is two contiguous
// 16-byte ds loads. A fragment is four global_load_b128 + packed cvt.
// M in {64,128} (template, k-loop fully unrolled). n multiple of 16.
// ACT: 0 none, 1 relu, 2 tanh, 3 sigmoid
// =====================================================================
template<int M, int ACT, bool BIAS, bool SCALE>
__global__ void gemm_wmma_k(const float* __restrict__ A, const float* __restrict__ W,
                            const float* __restrict__ bias, const float* __restrict__ rowscale,
                            float* __restrict__ out, int n)
{
    __shared__ _Float16 sWt[64 * M];           // transposed: sWt[c*M + k] = W[k*64 + c]
    for (int i = threadIdx.x; i < M * 64; i += blockDim.x) {
        int k = i >> 6, c = i & 63;
        sWt[c * M + k] = (_Float16)W[i];
    }
    __syncthreads();

    int wave = (int)((blockIdx.x * blockDim.x + threadIdx.x) >> 5);
    int lane = threadIdx.x & 31;
    int r0 = wave << 4;
    if (r0 >= n) return;                       // wave-uniform exit (after barrier)
    int lg = lane >> 4;                        // lane group 0/1
    int lm = lane & 15;

    v8f acc[4] = {};
    const float* Arow = A + (size_t)(r0 + lm) * M;

#pragma unroll
    for (int kk = 0; kk < M; kk += 32) {
        // A fragment (16-bit A 16x32 layout, ISA 7.12.2):
        //   halves 0..7  <- Arow[kk + lg*8 + 0..7]
        //   halves 8..15 <- Arow[kk + 16 + lg*8 + 0..7]
        const float4 a0 = *(const float4*)(Arow + kk + lg * 8);
        const float4 a1 = *(const float4*)(Arow + kk + lg * 8 + 4);
        const float4 a2 = *(const float4*)(Arow + kk + 16 + lg * 8);
        const float4 a3 = *(const float4*)(Arow + kk + 16 + lg * 8 + 4);
        v16h a;
        a[0]  = (_Float16)a0.x; a[1]  = (_Float16)a0.y; a[2]  = (_Float16)a0.z; a[3]  = (_Float16)a0.w;
        a[4]  = (_Float16)a1.x; a[5]  = (_Float16)a1.y; a[6]  = (_Float16)a1.z; a[7]  = (_Float16)a1.w;
        a[8]  = (_Float16)a2.x; a[9]  = (_Float16)a2.y; a[10] = (_Float16)a2.z; a[11] = (_Float16)a2.w;
        a[12] = (_Float16)a3.x; a[13] = (_Float16)a3.y; a[14] = (_Float16)a3.z; a[15] = (_Float16)a3.w;

#pragma unroll
        for (int c = 0; c < 4; ++c) {
            // B fragment for column tile c: column = c*16+lm, same K striping as A
            const _Float16* bp = &sWt[(c * 16 + lm) * M + kk + lg * 8];
            const v8h blo = *(const v8h*)(bp);
            const v8h bhi = *(const v8h*)(bp + 16);
            v16h b;
#pragma unroll
            for (int i = 0; i < 8; ++i) { b[i] = blo[i]; b[8 + i] = bhi[i]; }
            acc[c] = __builtin_amdgcn_wmma_f32_16x16x32_f16(false, a, false, b,
                                                            (short)0, acc[c], false, false);
        }
    }

    // epilogue: C/D layout -> VGPR r holds row r0 + r + 8*lg, col = c*16+lm
#pragma unroll
    for (int r = 0; r < 8; ++r) {
        int row = r0 + r + (lg << 3);
        float rs = SCALE ? rowscale[row] : 1.0f;
        float* orow = out + (size_t)row * 64;
#pragma unroll
        for (int c = 0; c < 4; ++c) {
            float v = acc[c][r];
            if (SCALE) v *= rs;
            if (BIAS)  v += bias[c * 16 + lm];
            if      (ACT == 1) v = fmaxf(v, 0.0f);
            else if (ACT == 2) v = tanhf(v);
            else if (ACT == 3) v = 1.0f / (1.0f + expf(-v));
            orow[c * 16 + lm] = v;
        }
    }
}

// ---------------- helpers / elementwise kernels ----------------
__global__ void fill_u32(unsigned* __restrict__ p, unsigned val, size_t count) {
    size_t i = (size_t)blockIdx.x * blockDim.x + threadIdx.x;
    if (i < count) p[i] = val;
}

// s_src/s_dst: per (node, head) dot of hp[n,h,:] with a[h,:]
__global__ void head_scores(const float* __restrict__ hp, const float* __restrict__ asrc,
                            const float* __restrict__ adst, float* __restrict__ ss,
                            float* __restrict__ sd, int n)
{
    int t = blockIdx.x * blockDim.x + threadIdx.x;
    if (t >= n * NH_H) return;
    int node = t / NH_H, h = t % NH_H;
    const float* hpp = hp + (size_t)node * HF_F + h * HD_D;
    float s1 = 0.f, s2 = 0.f;
#pragma unroll
    for (int d = 0; d < HD_D; ++d) {
        s1 += hpp[d] * asrc[h * HD_D + d];
        s2 += hpp[d] * adst[h * HD_D + d];
    }
    ss[t] = s1; sd[t] = s2;
}

// e = tanh(ss[src] + sd[dst]); store e; atomicMax key into mkey[dst,h]
__global__ void edge_e(const float* __restrict__ ss, const float* __restrict__ sd,
                       const int* __restrict__ esrc, const int* __restrict__ edst,
                       float* __restrict__ ebuf, unsigned* __restrict__ mkey, int E)
{
    int t = blockIdx.x * blockDim.x + threadIdx.x;
    if (t >= E * NH_H) return;
    int e = t >> 3, h = t & 7;
    int s = esrc[e], d = edst[e];
    float v = tanhf(ss[s * NH_H + h] + sd[d * NH_H + h]);
    ebuf[t] = v;
    atomicMax(&mkey[d * NH_H + h], fenc(v));
}

// decode max-keys in place to float; -inf -> 0
__global__ void key_finalize(unsigned* __restrict__ key, size_t count) {
    size_t i = (size_t)blockIdx.x * blockDim.x + threadIdx.x;
    if (i >= count) return;
    float v = fdec(key[i]);
    if (!isfinite(v)) v = 0.0f;
    ((float*)key)[i] = v;
}

// wgt = exp(e - m[dst]); overwrite ebuf; atomicAdd zsum[dst,h]
__global__ void edge_wgt(float* __restrict__ ebuf, const float* __restrict__ m,
                         const int* __restrict__ edst, float* __restrict__ zsum, int E)
{
    int t = blockIdx.x * blockDim.x + threadIdx.x;
    if (t >= E * NH_H) return;
    int e = t >> 3, h = t & 7;
    int d = edst[e];
    float w = expf(ebuf[t] - m[d * NH_H + h]);
    ebuf[t] = w;
    atomicAdd(&zsum[d * NH_H + h], w);
}

// agg[dst,h,:] += hp[src,h,:] * alpha
__global__ void edge_agg(const float* __restrict__ wgt, const float* __restrict__ zsum,
                         const float* __restrict__ hp, const int* __restrict__ esrc,
                         const int* __restrict__ edst, float* __restrict__ agg, int E)
{
    int t = blockIdx.x * blockDim.x + threadIdx.x;
    if (t >= E * NH_H) return;
    int e = t >> 3, h = t & 7;
    int s = esrc[e], d = edst[e];
    float alpha = wgt[t] / (zsum[d * NH_H + h] + 1e-9f);
    const float* hs = hp + (size_t)s * HF_F + h * HD_D;
    float* ad = agg + (size_t)d * HF_F + h * HD_D;
#pragma unroll
    for (int k = 0; k < HD_D; ++k) atomicAdd(&ad[k], hs[k] * alpha);
}

__global__ void relu_res(float* __restrict__ h, const float* __restrict__ agg, size_t count) {
    size_t i = (size_t)blockIdx.x * blockDim.x + threadIdx.x;
    if (i < count) h[i] = fmaxf(h[i] + agg[i], 0.0f);
}

__global__ void deg_kernel(const int* __restrict__ esrc, const int* __restrict__ edst,
                           float* __restrict__ dout, float* __restrict__ din, int E) {
    int t = blockIdx.x * blockDim.x + threadIdx.x;
    if (t >= E) return;
    atomicAdd(&dout[esrc[t]], 1.0f);
    atomicAdd(&din[edst[t]], 1.0f);
}

__global__ void rnorm_kernel(float* __restrict__ r, int n) {
    int i = blockIdx.x * blockDim.x + threadIdx.x;
    if (i < n) r[i] = rsqrtf(fmaxf(r[i], 1.0f));
}

// agg2[dst,f] += hs[src,f]
__global__ void scatter_add(const float* __restrict__ hs, const int* __restrict__ esrc,
                            const int* __restrict__ edst, float* __restrict__ agg, int E) {
    long long t = (long long)blockIdx.x * blockDim.x + threadIdx.x;
    if (t >= (long long)E * HF_F) return;
    int e = (int)(t >> 6), f = (int)(t & 63);
    atomicAdd(&agg[(size_t)edst[e] * HF_F + f], hs[(size_t)esrc[e] * HF_F + f]);
}

__global__ void gconv_epi(const float* __restrict__ agg, const float* __restrict__ rin,
                          const float* __restrict__ b, float* __restrict__ g, int n) {
    long long t = (long long)blockIdx.x * blockDim.x + threadIdx.x;
    if (t >= (long long)n * HF_F) return;
    int node = (int)(t >> 6), f = (int)(t & 63);
    g[t] = fmaxf(agg[t] * rin[node] + b[f], 0.0f);
}

// EdgeConv: msg = xt[src]-xt[dst]+xp[dst]+btheta+bphi ; atomicMax key
__global__ void edge_max(const float* __restrict__ xt, const float* __restrict__ xp,
                         const float* __restrict__ bt, const float* __restrict__ bp,
                         const int* __restrict__ esrc, const int* __restrict__ edst,
                         unsigned* __restrict__ fkey, int E) {
    long long t = (long long)blockIdx.x * blockDim.x + threadIdx.x;
    if (t >= (long long)E * HF_F) return;
    int e = (int)(t >> 6), f = (int)(t & 63);
    int s = esrc[e], d = edst[e];
    float msg = xt[(size_t)s * HF_F + f] - xt[(size_t)d * HF_F + f]
              + xp[(size_t)d * HF_F + f] + bt[f] + bp[f];
    atomicMax(&fkey[(size_t)d * HF_F + f], fenc(msg));
}

__global__ void concat_k(const float* __restrict__ a, const float* __restrict__ b,
                         float* __restrict__ cat, int n) {
    long long t = (long long)blockIdx.x * blockDim.x + threadIdx.x;
    if (t >= (long long)n * 2 * HF_F) return;
    int node = (int)(t >> 7), j = (int)(t & 127);
    cat[t] = (j < HF_F) ? a[(size_t)node * HF_F + j] : b[(size_t)node * HF_F + (j - HF_F)];
}

__global__ void fuse_k(const float* __restrict__ gate, const float* __restrict__ outp,
                       const float* __restrict__ fbr, float* __restrict__ fused, size_t count) {
    size_t i = (size_t)blockIdx.x * blockDim.x + threadIdx.x;
    if (i >= count) return;
    float g = gate[i];
    fused[i] = g * outp[i] + (1.0f - g) * fbr[i];
}

__global__ void final_head(const float* __restrict__ t, const float* __restrict__ lbW,
                           const float* __restrict__ lbb, float* __restrict__ out, int n) {
    int i = blockIdx.x * blockDim.x + threadIdx.x;
    if (i >= n) return;
    const float* tr = t + (size_t)i * HF_F;
    float a0 = lbb[0], a1 = lbb[1];
#pragma unroll
    for (int f = 0; f < HF_F; ++f) { float v = tr[f]; a0 += v * lbW[2 * f]; a1 += v * lbW[2 * f + 1]; }
    out[2 * i] = a0; out[2 * i + 1] = a1;
}

// ---------------- host-side launch ----------------
static inline unsigned blocksFor(long long n, int bs) { return (unsigned)((n + bs - 1) / bs); }

// act: 0 none, 1 relu, 2 tanh, 3 sigmoid
static void launch_gemm(const float* A, const float* W, const float* bias,
                        const float* rowscale, float* out, int n, int M,
                        int act, hipStream_t s) {
    int waves = n / 16;                         // n is a multiple of 16 (50000)
    dim3 grid(blocksFor(waves, 8)), blk(256);   // 8 waves / block
    if (M == 128) {
        if (rowscale)        hipLaunchKernelGGL((gemm_wmma_k<128, 0, false, true >), grid, blk, 0, s, A, W, bias, rowscale, out, n);
        else if (bias)       hipLaunchKernelGGL((gemm_wmma_k<128, 0, true,  false>), grid, blk, 0, s, A, W, bias, rowscale, out, n);
        else                 hipLaunchKernelGGL((gemm_wmma_k<128, 0, false, false>), grid, blk, 0, s, A, W, bias, rowscale, out, n);
    } else {
        if (rowscale)        hipLaunchKernelGGL((gemm_wmma_k<64, 0, false, true >), grid, blk, 0, s, A, W, bias, rowscale, out, n);
        else if (act == 1)   hipLaunchKernelGGL((gemm_wmma_k<64, 1, true,  false>), grid, blk, 0, s, A, W, bias, rowscale, out, n);
        else if (act == 2)   hipLaunchKernelGGL((gemm_wmma_k<64, 2, true,  false>), grid, blk, 0, s, A, W, bias, rowscale, out, n);
        else if (act == 3)   hipLaunchKernelGGL((gemm_wmma_k<64, 3, true,  false>), grid, blk, 0, s, A, W, bias, rowscale, out, n);
        else                 hipLaunchKernelGGL((gemm_wmma_k<64, 0, true,  false>), grid, blk, 0, s, A, W, bias, rowscale, out, n);
    }
}

extern "C" void kernel_launch(void* const* d_in, const int* in_sizes, int n_in,
                              void* d_out, int out_size, void* d_ws, size_t ws_size,
                              hipStream_t stream) {
    const float* x        = (const float*)d_in[0];
    const int*   esrc     = (const int*)d_in[1];
    const int*   edst     = (const int*)d_in[2];
    const float* dec_W0   = (const float*)d_in[3];
    const float* dec_b0   = (const float*)d_in[4];
    const float* dec_W    = (const float*)d_in[5];
    const float* dec_bW   = (const float*)d_in[6];
    const float* dec_asrc = (const float*)d_in[7];
    const float* dec_adst = (const float*)d_in[8];
    const float* gcn_W1   = (const float*)d_in[9];
    const float* gcn_b1   = (const float*)d_in[10];
    const float* gcn_W2   = (const float*)d_in[11];
    const float* gcn_b2   = (const float*)d_in[12];
    const float* gcn_mW1  = (const float*)d_in[13];
    const float* gcn_mb1  = (const float*)d_in[14];
    const float* gcn_mW2  = (const float*)d_in[15];
    const float* gcn_mb2  = (const float*)d_in[16];
    const float* ec_theta = (const float*)d_in[17];
    const float* ec_bth   = (const float*)d_in[18];
    const float* ec_phi   = (const float*)d_in[19];
    const float* ec_bph   = (const float*)d_in[20];
    const float* lin1_W   = (const float*)d_in[21];
    const float* lin1_b   = (const float*)d_in[22];
    const float* gate_W   = (const float*)d_in[23];
    const float* gate_b   = (const float*)d_in[24];
    const float* la_W     = (const float*)d_in[25];
    const float* la_b     = (const float*)d_in[26];
    const float* lb_W     = (const float*)d_in[27];
    const float* lb_b     = (const float*)d_in[28];
    float* out = (float*)d_out;

    const int N = in_sizes[0] / IN_F;
    const int E = in_sizes[1];
    const size_t NF = (size_t)N * HF_F;
    const size_t NH8 = (size_t)N * NH_H;
    const size_t EH = (size_t)E * NH_H;      // also >= N*2*HF here

    float* ws = (float*)d_ws;
    float* h    = ws;              // out1 (persists)
    float* hp   = ws + 1 * NF;     // decoder hp; later xt
    float* agg  = ws + 2 * NF;     // decoder agg; later GCN agg2
    float* tmp  = ws + 3 * NF;     // hs / MLP hidden / fused
    float* gbuf = ws + 4 * NF;     // GCN g / tanh buffer
    float* out2 = ws + 5 * NF;
    float* xp   = ws + 6 * NF;
    float* fbr  = ws + 7 * NF;     // Fkey (u32) then Fbr (f32)
    float* outp = ws + 8 * NF;
    float* gate = ws + 9 * NF;
    float* ss   = ws + 10 * NF;
    float* sd   = ss + NH8;
    float* mkey = sd + NH8;        // u32 keys then float m
    float* zsum = mkey + NH8;
    float* rout = zsum + NH8;
    float* rin  = rout + N;
    float* ebuf = rin + N;         // E*NH floats; also the concat buffer (N*128)

    const int BS = 256;

    // ---- Decoder branch ----
    launch_gemm(x, dec_W0, dec_b0, nullptr, h, N, IN_F, 0, stream);
    for (int l = 0; l < L_LAYERS; ++l) {
        launch_gemm(h, dec_W + (size_t)l * HF_F * HF_F, dec_bW + (size_t)l * HF_F,
                    nullptr, hp, N, HF_F, 0, stream);
        hipLaunchKernelGGL(head_scores, dim3(blocksFor((long long)N * NH_H, BS)), dim3(BS), 0, stream,
                           hp, dec_asrc + l * NH_H * HD_D, dec_adst + l * NH_H * HD_D, ss, sd, N);
        hipLaunchKernelGGL(fill_u32, dim3(blocksFor(NH8, BS)), dim3(BS), 0, stream,
                           (unsigned*)mkey, KEY_NEG_INF, NH8);
        hipLaunchKernelGGL(fill_u32, dim3(blocksFor(NH8, BS)), dim3(BS), 0, stream,
                           (unsigned*)zsum, 0u, NH8);
        hipLaunchKernelGGL(fill_u32, dim3(blocksFor(NF, BS)), dim3(BS), 0, stream,
                           (unsigned*)agg, 0u, NF);
        hipLaunchKernelGGL(edge_e, dim3(blocksFor(EH, BS)), dim3(BS), 0, stream,
                           ss, sd, esrc, edst, ebuf, (unsigned*)mkey, E);
        hipLaunchKernelGGL(key_finalize, dim3(blocksFor(NH8, BS)), dim3(BS), 0, stream,
                           (unsigned*)mkey, NH8);
        hipLaunchKernelGGL(edge_wgt, dim3(blocksFor(EH, BS)), dim3(BS), 0, stream,
                           ebuf, mkey, edst, zsum, E);
        hipLaunchKernelGGL(edge_agg, dim3(blocksFor(EH, BS)), dim3(BS), 0, stream,
                           ebuf, zsum, hp, esrc, edst, agg, E);
        hipLaunchKernelGGL(relu_res, dim3(blocksFor(NF, BS)), dim3(BS), 0, stream, h, agg, NF);
    }

    // ---- GCN branch ----
    hipLaunchKernelGGL(fill_u32, dim3(blocksFor(N, BS)), dim3(BS), 0, stream, (unsigned*)rout, 0u, (size_t)N);
    hipLaunchKernelGGL(fill_u32, dim3(blocksFor(N, BS)), dim3(BS), 0, stream, (unsigned*)rin, 0u, (size_t)N);
    hipLaunchKernelGGL(deg_kernel, dim3(blocksFor(E, BS)), dim3(BS), 0, stream, esrc, edst, rout, rin, E);
    hipLaunchKernelGGL(rnorm_kernel, dim3(blocksFor(N, BS)), dim3(BS), 0, stream, rout, N);
    hipLaunchKernelGGL(rnorm_kernel, dim3(blocksFor(N, BS)), dim3(BS), 0, stream, rin, N);

    launch_gemm(x, gcn_W1, nullptr, rout, tmp, N, IN_F, 0, stream);
    hipLaunchKernelGGL(fill_u32, dim3(blocksFor(NF, BS)), dim3(BS), 0, stream, (unsigned*)agg, 0u, NF);
    hipLaunchKernelGGL(scatter_add, dim3(blocksFor((long long)E * HF_F, BS)), dim3(BS), 0, stream,
                       tmp, esrc, edst, agg, E);
    hipLaunchKernelGGL(gconv_epi, dim3(blocksFor(NF, BS)), dim3(BS), 0, stream, agg, rin, gcn_b1, gbuf, N);

    launch_gemm(gbuf, gcn_W2, nullptr, rout, tmp, N, HF_F, 0, stream);
    hipLaunchKernelGGL(fill_u32, dim3(blocksFor(NF, BS)), dim3(BS), 0, stream, (unsigned*)agg, 0u, NF);
    hipLaunchKernelGGL(scatter_add, dim3(blocksFor((long long)E * HF_F, BS)), dim3(BS), 0, stream,
                       tmp, esrc, edst, agg, E);
    hipLaunchKernelGGL(gconv_epi, dim3(blocksFor(NF, BS)), dim3(BS), 0, stream, agg, rin, gcn_b2, gbuf, N);

    launch_gemm(gbuf, gcn_mW1, gcn_mb1, nullptr, tmp, N, HF_F, 1, stream);
    launch_gemm(tmp, gcn_mW2, gcn_mb2, nullptr, out2, N, HF_F, 0, stream);

    // ---- EdgeConv branch ----
    launch_gemm(x, ec_theta, nullptr, nullptr, hp, N, IN_F, 0, stream);   // xt
    launch_gemm(x, ec_phi,   nullptr, nullptr, xp, N, IN_F, 0, stream);   // xp
    hipLaunchKernelGGL(fill_u32, dim3(blocksFor(NF, BS)), dim3(BS), 0, stream,
                       (unsigned*)fbr, KEY_NEG_INF, NF);
    hipLaunchKernelGGL(edge_max, dim3(blocksFor((long long)E * HF_F, BS)), dim3(BS), 0, stream,
                       hp, xp, ec_bth, ec_bph, esrc, edst, (unsigned*)fbr, E);
    hipLaunchKernelGGL(key_finalize, dim3(blocksFor(NF, BS)), dim3(BS), 0, stream, (unsigned*)fbr, NF);

    // ---- gated fusion + output head ----
    hipLaunchKernelGGL(concat_k, dim3(blocksFor((long long)N * 2 * HF_F, BS)), dim3(BS), 0, stream,
                       h, out2, ebuf, N);
    launch_gemm(ebuf, lin1_W, lin1_b, nullptr, outp, N, 2 * HF_F, 0, stream);
    launch_gemm(outp, gate_W, gate_b, nullptr, gate, N, HF_F, 3, stream);
    hipLaunchKernelGGL(fuse_k, dim3(blocksFor(NF, BS)), dim3(BS), 0, stream, gate, outp, fbr, tmp, NF);
    launch_gemm(tmp, la_W, la_b, nullptr, gbuf, N, HF_F, 2, stream);
    hipLaunchKernelGGL(final_head, dim3(blocksFor(N, BS)), dim3(BS), 0, stream, gbuf, lb_W, lb_b, out, N);

    (void)n_in; (void)out_size; (void)ws_size;
}
